// SkeletonGNN_71004399338036
// MI455X (gfx1250) — matrix operand order repaired and checked by reading, compiled, not run on hardware
//
#include <hip/hip_runtime.h>

// ---------------------------------------------------------------------------
// SkeletonGNN on MI455X (gfx1250, wave32, WMMA + TDM).
// rows = B*J rows of D=64 features. Three per-row GEMMs done with
// v_wmma_f32_16x16x32_f16 (f16 operands, f32 accumulate, K=64 -> 2 wmma/tile).
// Block = 256 threads = 8 waves; each wave owns a 16-row M-tile.
// Bulk contiguous copies (f16 weights, block's h tile) use TENSOR_LOAD_TO_LDS.
// ---------------------------------------------------------------------------

#define NUM_J 33
#define FEAT 64
#define ROWS_PER_BLOCK 128

typedef __attribute__((ext_vector_type(16))) _Float16 v16h;
typedef __attribute__((ext_vector_type(8)))  float    v8f;
typedef __attribute__((ext_vector_type(4)))  unsigned int v4u;
typedef __attribute__((ext_vector_type(8)))  int      v8i;
typedef __attribute__((ext_vector_type(4)))  int      v4i;

union Frag16 { v16h v; uint4 q[2]; };

// LDS partition byte offsets (dynamic LDS region starts at 0).
#define LDS_W_OFF    0u                                   // 28672 halves
#define LDS_H16_OFF  (28672u * 2u)                        // 8192 halves
#define LDS_AGG_OFF  (LDS_H16_OFF + 8192u * 2u)
#define LDS_MSG_OFF  (LDS_AGG_OFF + 8192u * 2u)
#define LDS_H32_OFF  (LDS_MSG_OFF + 8192u * 2u)           // 8192 floats
#define LDS_TOTAL    (LDS_H32_OFF + 8192u * 4u)           // 139,264 B

// Neighbor tables from EDGES (self first; padded with 0 / cnt covers valid).
__device__ __constant__ int NBR_IDX_C[NUM_J][4] = {
    {0,0,0,0},{1,0,0,0},{2,0,0,0},{3,0,0,0},{4,0,0,0},{5,0,0,0},{6,0,0,0},
    {7,0,0,0},{8,0,0,0},{9,0,0,0},{10,0,0,0},
    {11,12,23,13},{12,11,24,14},{13,11,15,0},{14,12,16,0},{15,13,0,0},{16,14,0,0},
    {17,0,0,0},{18,0,0,0},{19,0,0,0},{20,0,0,0},{21,0,0,0},{22,0,0,0},
    {23,11,24,25},{24,12,23,26},{25,23,27,0},{26,24,28,0},{27,25,0,0},{28,26,0,0},
    {29,0,0,0},{30,0,0,0},{31,0,0,0},{32,0,0,0}};
__device__ __constant__ int NBR_CNT_C[NUM_J] = {
    1,1,1,1,1,1,1,1,1,1,1, 4,4,3,3,2,2, 1,1,1,1,1,1, 4,4,3,3,2,2, 1,1,1,1};

__device__ __forceinline__ v8f wmma16(v16h a, v16h b, v8f c) {
    // D = A(16x32 f16) * B(32x16 f16) + C(16x16 f32)
    return __builtin_amdgcn_wmma_f32_16x16x32_f16(
        /*neg_a=*/false, a, /*neg_b=*/false, b,
        /*c_mod=*/(short)0, c, /*reuse_a=*/false, /*reuse_b=*/false);
}

// A operand: row-major f16 tile (stride 64). Lane ln holds M=ln.
// halves[i] = K(kofs + lhi*16 + i), i = 0..15
__device__ __forceinline__ v16h load_frag_A(const _Float16* base, int ln, int lhi, int kofs) {
    Frag16 f;
    const _Float16* p = base + ln * FEAT + kofs + lhi * 16;
    f.q[0] = *(const uint4*)(p);
    f.q[1] = *(const uint4*)(p + 8);
    return f.v;
}

// B operand: weight stored row-major as W[n][k] (== original fp32 layout,
// since out[.,n] = sum_k in[.,k] * W[n*64+k]). Lane ln holds column N=ncol;
// halves[i] = K(kt*32 + lhi*16 + i), i = 0..15 -> 32 contiguous bytes.
__device__ __forceinline__ v16h load_frag_B(const _Float16* w, int ncol, int lhi, int kt) {
    Frag16 f;
    const uint4* p = (const uint4*)(w + ncol * FEAT + kt * 32 + lhi * 16);
    f.q[0] = p[0];
    f.q[1] = p[1];
    return f.v;
}

__device__ __forceinline__ float sigmoid_f(float x) {
    return 1.0f / (1.0f + __expf(-x));
}
__device__ __forceinline__ float tanh_f(float x) {
    float e = __expf(2.0f * x);
    return (e - 1.0f) / (e + 1.0f);
}

// ---------------------------------------------------------------------------
// TDM: 1-D contiguous copy global -> LDS, n8 elements of 8 bytes.
// D# built per CDNA5 ISA:
//   group0: count=1 | lds_addr | global_addr[56:0] | type=2
//   group1: data_size=3 (8B), tensor_dim0 = tile_dim0 = n8 (no OOB),
//           tensor_dim1 = 1, tile_dim1/2 = 0 (1-D), strides benign.
//   remaining groups: zero (dims 2..4 unused).
// This toolchain's builtin takes 6 args (…, int32x4, int32x4, int32x8, cpol).
// ---------------------------------------------------------------------------
__device__ __forceinline__ void tdm_load_1d(unsigned lds_byte_off,
                                            const void* gptr, unsigned n8) {
    unsigned long long ga = (unsigned long long)(uintptr_t)gptr;
    v4u g0;
    g0.x = 1u;                                          // count=1, user mode
    g0.y = lds_byte_off;                                // lds_addr
    g0.z = (unsigned)(ga & 0xffffffffull);              // global_addr lo
    g0.w = (unsigned)((ga >> 32) & 0x01ffffffull) | 0x80000000u;  // hi | type=2
    v8i g1;
    g1[0] = 0x00030000;                                 // data_size = 3 (8B)
    g1[1] = (int)((n8 & 0xffffu) << 16);                // tensor_dim0 lo
    g1[2] = (int)((n8 >> 16) | (1u << 16));             // tensor_dim0 hi | tensor_dim1=1
    g1[3] = (int)(n8 << 16);                            // tile_dim0 (16-bit)
    g1[4] = 0;                                          // tile_dim1/2 = 0 (1-D)
    g1[5] = (int)n8;                                    // tensor_dim0_stride lo
    g1[6] = 0;
    g1[7] = 0;
    v4i z4 = {0, 0, 0, 0};
    v8i z8 = {0, 0, 0, 0, 0, 0, 0, 0};
    __builtin_amdgcn_tensor_load_to_lds(g0, g1, z4, z4, z8, 0);
}

// ---------------------------------------------------------------------------
// Prologue: fp32 weights -> f16 workspace. Layout (halves):
//   [0,4096)       msg_w  (64 x 64)
//   [4096,16384)   w_ih   (192 x 64)
//   [16384,28672)  w_hh   (192 x 64)
// ---------------------------------------------------------------------------
__global__ void cvt_weights_kernel(const float* __restrict__ msg_w,
                                   const float* __restrict__ w_ih,
                                   const float* __restrict__ w_hh,
                                   _Float16* __restrict__ ws) {
    int i = blockIdx.x * 256 + threadIdx.x;
    if (i >= 28672) return;
    float v;
    if (i < 4096)        v = msg_w[i];
    else if (i < 16384)  v = w_ih[i - 4096];
    else                 v = w_hh[i - 16384];
    ws[i] = (_Float16)v;
}

// ---------------------------------------------------------------------------
// Main kernel.
// ---------------------------------------------------------------------------
__global__ __launch_bounds__(256)
void skel_gnn_kernel(const float* __restrict__ feats,
                     const float* __restrict__ vis,
                     const float* __restrict__ msg_b,
                     const float* __restrict__ b_ih,
                     const float* __restrict__ b_hh,
                     const _Float16* __restrict__ w16,
                     float* __restrict__ out) {
    extern __shared__ __align__(16) char smem[];
    _Float16* sW   = (_Float16*)(smem + LDS_W_OFF);     // 28672 halves
    _Float16* sH16 = (_Float16*)(smem + LDS_H16_OFF);   // 128*64 halves
    _Float16* sAgg = (_Float16*)(smem + LDS_AGG_OFF);
    _Float16* sMsg = (_Float16*)(smem + LDS_MSG_OFF);
    float*    sH32 = (float*)   (smem + LDS_H32_OFF);   // 128*64 floats

    const int tid  = threadIdx.x;
    const int wave = tid >> 5;

    // ---- Phase A0: TDM async copies into LDS (issued once, by wave 0) ------
    if (wave == 0) {
        // f16 weights: 57,344 B = 7168 x 8B
        tdm_load_1d(LDS_W_OFF, w16, 28672u / 4u);
        // this block's h tile: 128 consecutive rows x 64 f32 = 32,768 B = 4096 x 8B
        tdm_load_1d(LDS_H32_OFF,
                    feats + (size_t)blockIdx.x * ROWS_PER_BLOCK * FEAT,
                    (ROWS_PER_BLOCK * FEAT) / 2u);
        __builtin_amdgcn_s_wait_tensorcnt(0);
    }
    __syncthreads();

    // ---- Phase A1: neighbor aggregation + stage h (f16) and agg (f16) ------
    {
        const int row_local = tid >> 1;               // 128 rows, 2 threads/row
        const int half      = tid & 1;                // 32 cols each
        const int r  = blockIdx.x * ROWS_PER_BLOCK + row_local;
        const int b  = r / NUM_J;
        const int j  = r - b * NUM_J;
        const int cnt = NBR_CNT_C[j];
        const float invc = 1.0f / (float)cnt;
        const int i1 = NBR_IDX_C[j][1], i2 = NBR_IDX_C[j][2], i3 = NBR_IDX_C[j][3];
        const float* vb = vis + (size_t)b * NUM_J;
        const float w0 = vb[j];
        const float w1 = (cnt > 1) ? vb[i1] : 0.0f;
        const float w2 = (cnt > 2) ? vb[i2] : 0.0f;
        const float w3 = (cnt > 3) ? vb[i3] : 0.0f;
        const float* fb = feats + (size_t)b * NUM_J * FEAT;
        const int colbase = half * 32;

        #pragma unroll
        for (int c = 0; c < 4; ++c) {
            const int col = colbase + c * 8;
            // self row comes from the TDM-staged LDS tile
            float4 s0 = *(const float4*)(sH32 + row_local * FEAT + col);
            float4 s1 = *(const float4*)(sH32 + row_local * FEAT + col + 4);
            _Float16* hp = sH16 + row_local * FEAT + col;
            hp[0]=(_Float16)s0.x; hp[1]=(_Float16)s0.y; hp[2]=(_Float16)s0.z; hp[3]=(_Float16)s0.w;
            hp[4]=(_Float16)s1.x; hp[5]=(_Float16)s1.y; hp[6]=(_Float16)s1.z; hp[7]=(_Float16)s1.w;
            float a0=w0*s0.x, a1=w0*s0.y, a2=w0*s0.z, a3=w0*s0.w;
            float a4=w0*s1.x, a5=w0*s1.y, a6=w0*s1.z, a7=w0*s1.w;
            if (cnt > 1) {
                float4 t0 = *(const float4*)(fb + i1 * FEAT + col);
                float4 t1 = *(const float4*)(fb + i1 * FEAT + col + 4);
                a0+=w1*t0.x; a1+=w1*t0.y; a2+=w1*t0.z; a3+=w1*t0.w;
                a4+=w1*t1.x; a5+=w1*t1.y; a6+=w1*t1.z; a7+=w1*t1.w;
            }
            if (cnt > 2) {
                float4 t0 = *(const float4*)(fb + i2 * FEAT + col);
                float4 t1 = *(const float4*)(fb + i2 * FEAT + col + 4);
                a0+=w2*t0.x; a1+=w2*t0.y; a2+=w2*t0.z; a3+=w2*t0.w;
                a4+=w2*t1.x; a5+=w2*t1.y; a6+=w2*t1.z; a7+=w2*t1.w;
            }
            if (cnt > 3) {
                float4 t0 = *(const float4*)(fb + i3 * FEAT + col);
                float4 t1 = *(const float4*)(fb + i3 * FEAT + col + 4);
                a0+=w3*t0.x; a1+=w3*t0.y; a2+=w3*t0.z; a3+=w3*t0.w;
                a4+=w3*t1.x; a5+=w3*t1.y; a6+=w3*t1.z; a7+=w3*t1.w;
            }
            _Float16* ap = sAgg + row_local * FEAT + col;
            ap[0]=(_Float16)(a0*invc); ap[1]=(_Float16)(a1*invc);
            ap[2]=(_Float16)(a2*invc); ap[3]=(_Float16)(a3*invc);
            ap[4]=(_Float16)(a4*invc); ap[5]=(_Float16)(a5*invc);
            ap[6]=(_Float16)(a6*invc); ap[7]=(_Float16)(a7*invc);
        }
    }
    __syncthreads();

    // ---- Phase B: per-wave WMMA over its 16-row tile ----
    const int lane = tid & 31;
    const int ln   = lane & 15;
    const int lhi  = lane >> 4;

    const _Float16* aggT = sAgg + wave * 16 * FEAT;
    const _Float16* hT   = sH16 + wave * 16 * FEAT;
    _Float16*       msgT = sMsg + wave * 16 * FEAT;
    const _Float16* Wmsg = sW;            // 64 rows
    const _Float16* Wih  = sW + 4096;     // 192 rows
    const _Float16* Whh  = sW + 16384;    // 192 rows

    // msgs = agg @ msg_w^T + msg_b   (8 wmma)
    {
        v16h a0 = load_frag_A(aggT, ln, lhi, 0);
        v16h a1 = load_frag_A(aggT, ln, lhi, 32);
        #pragma unroll
        for (int nt = 0; nt < 4; ++nt) {
            const int nc = nt * 16 + ln;
            v8f c = {};
            c = wmma16(a0, load_frag_B(Wmsg, nc, lhi, 0), c);
            c = wmma16(a1, load_frag_B(Wmsg, nc, lhi, 1), c);
            const float mb = msg_b[nc];
            #pragma unroll
            for (int v = 0; v < 8; ++v)
                msgT[(v + lhi * 8) * FEAT + nc] = (_Float16)(c[v] + mb);
        }
    }
    __syncthreads();

    // GRU gates: gi = msgs@w_ih^T + b_ih ; gh = h@w_hh^T + b_hh  (48 wmma)
    v16h aM0 = load_frag_A(msgT, ln, lhi, 0);
    v16h aM1 = load_frag_A(msgT, ln, lhi, 32);
    v16h aH0 = load_frag_A(hT,   ln, lhi, 0);
    v16h aH1 = load_frag_A(hT,   ln, lhi, 32);

    float rs[4][8], zs[4][8];
    const size_t outbase = ((size_t)blockIdx.x * ROWS_PER_BLOCK + wave * 16) * FEAT;

    #pragma unroll
    for (int g = 0; g < 3; ++g) {
        #pragma unroll
        for (int nt = 0; nt < 4; ++nt) {
            const int nc = g * 64 + nt * 16 + ln;   // column into 192-wide gates
            v8f gi = {};
            gi = wmma16(aM0, load_frag_B(Wih, nc, lhi, 0), gi);
            gi = wmma16(aM1, load_frag_B(Wih, nc, lhi, 1), gi);
            v8f gh = {};
            gh = wmma16(aH0, load_frag_B(Whh, nc, lhi, 0), gh);
            gh = wmma16(aH1, load_frag_B(Whh, nc, lhi, 1), gh);
            const float bi = b_ih[nc];
            const float bh = b_hh[nc];
            #pragma unroll
            for (int v = 0; v < 8; ++v) {
                const float giv = gi[v] + bi;
                const float ghv = gh[v] + bh;
                if (g == 0) {
                    rs[nt][v] = sigmoid_f(giv + ghv);
                } else if (g == 1) {
                    zs[nt][v] = sigmoid_f(giv + ghv);
                } else {
                    const int rl = wave * 16 + v + lhi * 8;
                    const float hv = sH32[rl * FEAT + nt * 16 + ln];
                    const float nn = tanh_f(giv + rs[nt][v] * ghv);
                    const float z  = zs[nt][v];
                    out[outbase + (size_t)(v + lhi * 8) * FEAT + nt * 16 + ln] =
                        (1.0f - z) * nn + z * hv;
                }
            }
        }
    }
}

// ---------------------------------------------------------------------------
extern "C" void kernel_launch(void* const* d_in, const int* in_sizes, int n_in,
                              void* d_out, int out_size, void* d_ws, size_t ws_size,
                              hipStream_t stream) {
    const float* feats = (const float*)d_in[0];  // (B,33,64)
    const float* vis   = (const float*)d_in[1];  // (B,33)
    const float* msg_w = (const float*)d_in[2];  // (64,64)
    const float* msg_b = (const float*)d_in[3];  // (64,)
    const float* w_ih  = (const float*)d_in[4];  // (192,64)
    const float* w_hh  = (const float*)d_in[5];  // (192,64)
    const float* b_ih  = (const float*)d_in[6];  // (192,)
    const float* b_hh  = (const float*)d_in[7];  // (192,)
    float* out = (float*)d_out;
    _Float16* w16 = (_Float16*)d_ws;             // 57,344 bytes used

    const int B = in_sizes[0] / (NUM_J * FEAT);  // 32768
    const int rows = B * NUM_J;                  // 1,081,344 = 8448 * 128
    const int blocks = rows / ROWS_PER_BLOCK;

    cvt_weights_kernel<<<(28672 + 255) / 256, 256, 0, stream>>>(msg_w, w_ih, w_hh, w16);

    skel_gnn_kernel<<<blocks, 256, LDS_TOTAL, stream>>>(
        feats, vis, msg_b, b_ih, b_hh, w16, out);
}
